// LearnedLandmarkBranch_61873298866644
// MI455X (gfx1250) — matrix operand order repaired and checked by reading, not compiled, not run on hardware
//
#include <hip/hip_runtime.h>
#include <hip/hip_bf16.h>

// ---------------------------------------------------------------------------
// Types for WMMA (CDNA5 gfx1250, wave32)
// ---------------------------------------------------------------------------
typedef __attribute__((ext_vector_type(16))) __bf16 v16bf;
typedef __attribute__((ext_vector_type(8)))  float  v8f;
typedef __attribute__((ext_vector_type(4)))  unsigned int v4u;
typedef __attribute__((ext_vector_type(4)))  float  v4f;

#define TAUf 0.03f
#define EPSf 1e-6f

// Problem dims
constexpr int Bn  = 32;
constexpr int Cin = 1024;
constexpr int Hh  = 28;
constexpr int Ww  = 28;
constexpr int HW  = 784;
constexpr int C1  = 256;   // conv1 out channels
constexpr int Kk  = 12;    // keypoints
constexpr int Npix  = Bn * HW;        // 25088
constexpr int NTILE = 64;
constexpr int NBLK  = Npix / NTILE;   // 392 (exact)
constexpr int KSTEPS = 9 * (Cin / 32); // 288 K-steps of 32

// Workspace layout (byte offsets)
constexpr size_t WS_W1 = 0;                                 // 9*256*1024 bf16
constexpr size_t WS_W2 = WS_W1 + (size_t)9 * C1 * Cin * 2;  // 16*256 bf16
constexpr size_t WS_LG = WS_W2 + (size_t)16 * 256 * 2;      // 32*12*784 f32
constexpr size_t WS_PB = WS_LG + (size_t)Bn * Kk * HW * 4;  // 32*16*800 bf16
constexpr size_t WS_PT = WS_PB + (size_t)Bn * 16 * 800 * 2; // 64 f32 partials

// d_out layout (float offsets), outputs concatenated in return order
constexpr int OUT_PROBS = 0;                       // [32,12,28,28]
constexpr int OUT_COORD = Bn * Kk * HW;            // [32,12,2]
constexpr int OUT_FEATK = OUT_COORD + Bn * Kk * 2; // [32, 12*1024]
constexpr int OUT_DIV   = OUT_FEATK + Bn * Kk * Cin;
constexpr int OUT_ENT   = OUT_DIV + 1;

// ---------------------------------------------------------------------------
// CDNA5 async copy: global -> LDS (ASYNCcnt tracked), 16B per lane
// ---------------------------------------------------------------------------
__device__ __forceinline__ void async_b128(const void* gsrc, const void* ldst) {
  unsigned loff = (unsigned)(unsigned long long)ldst;       // flat LDS addr[31:0] == LDS offset
  unsigned long long ga = (unsigned long long)gsrc;
  asm volatile("global_load_async_to_lds_b128 %0, %1, off"
               :: "v"(loff), "v"(ga) : "memory");
}
__device__ __forceinline__ void wait_async0() {
  asm volatile("s_wait_asynccnt 0x0" ::: "memory");
}

// ---------------------------------------------------------------------------
// Kernel 1a: W1 fp32 [O,C,3,3] -> bf16 [rs][O][C]
// ---------------------------------------------------------------------------
__global__ void prep_w1(const float* __restrict__ W1, __bf16* __restrict__ A) {
  int idx = blockIdx.x * 256 + threadIdx.x;
  const int total = 9 * C1 * Cin;
  if (idx >= total) return;
  int rs  = idx / (C1 * Cin);
  int rem = idx - rs * (C1 * Cin);
  int o = rem / Cin;
  int c = rem - o * Cin;
  A[idx] = (__bf16)W1[(size_t)(o * Cin + c) * 9 + rs];
}

// Kernel 1b: W2 fp32 [12,256] -> bf16 [16][256] (rows 12..15 zero)
__global__ void prep_w2(const float* __restrict__ W2, __bf16* __restrict__ A2) {
  int idx = blockIdx.x * 256 + threadIdx.x;
  if (idx >= 16 * 256) return;
  int m = idx >> 8, c = idx & 255;
  A2[idx] = (m < Kk) ? (__bf16)W2[m * 256 + c] : (__bf16)0.0f;
}

// ---------------------------------------------------------------------------
// Kernel 2: conv1 (implicit GEMM, bf16 WMMA, double-buffered + async A copy)
//           + fused bias/ReLU + conv2 (WMMA). 256 out-ch x 64 px per WG.
// LDS map: [0,20480) As buf0 | [20480,40960) As buf1
//          [40960,46080) Bs buf0 | [46080,51200) Bs buf1
// Epilogue aliases [0,33792) as Ys[64][264] bf16.
// ---------------------------------------------------------------------------
__global__ __launch_bounds__(256) void conv_fused(
    const float* __restrict__ feat, const __bf16* __restrict__ Abf,
    const __bf16* __restrict__ W2bf, const float* __restrict__ b1,
    const float* __restrict__ b2, float* __restrict__ logits_out)
{
  __shared__ __align__(16) char smem[51200];

  const int tid    = threadIdx.x;
  const int lane   = tid & 31;
  const int wv     = tid >> 5;   // 8 waves
  const int n_sub  = wv & 3;     // 16-pixel column block
  const int m_half = wv >> 2;    // half of the 16 M-subtiles
  const int n0     = blockIdx.x * NTILE;

  // Per-thread pixel decode for the im2col gather of B
  const int j     = tid & 63;    // pixel within tile
  const int cbase = (tid >> 6) * 8;  // 8 contiguous channels per thread
  const int p     = n0 + j;
  const int bimg  = p / HW;
  const int rem   = p - bimg * HW;
  const int py    = rem / Ww;
  const int px    = rem - py * Ww;
  const float* featb = feat + (size_t)bimg * Cin * HW;

  const int fm    = lane & 15;
  const int khalf = lane >> 4;

  v8f acc[8] = {};
  float bregs[8];
  float bmask = 0.0f;
  int   boff  = 0;

  // ---- staging helpers ----
  auto stageA = [&](int step, int buf) {
    const __bf16* Ars = Abf + (size_t)(step >> 5) * (C1 * Cin) + ((step & 31) << 5);
    char* dst = smem + buf * 20480;
    #pragma unroll
    for (int pass = 0; pass < 4; ++pass) {
      int li   = pass * 256 + tid;
      int o    = li >> 2;
      int coff = (li & 3) * 8;
      async_b128(Ars + (size_t)o * Cin + coff, dst + o * 80 + coff * 2);
    }
  };
  auto loadB = [&](int step) {
    int rs = step >> 5;
    int c0 = (step & 31) << 5;
    int r = rs / 3, s = rs - 3 * (rs / 3);
    int yy = py + r - 1, xx = px + s - 1;
    bool valid = (yy >= 0) && (yy < Hh) && (xx >= 0) && (xx < Ww);
    bmask = valid ? 1.0f : 0.0f;
    int yc = min(max(yy, 0), Hh - 1);
    int xc = min(max(xx, 0), Ww - 1);
    boff = yc * Ww + xc;
    const float* g = featb + (size_t)(c0 + cbase) * HW + boff;
    #pragma unroll
    for (int e = 0; e < 8; ++e) bregs[e] = g[(size_t)e * HW];   // branchless, clamped
  };
  auto storeB = [&](int buf) {
    union { __bf16 h[8]; v4u q; } pk;
    #pragma unroll
    for (int e = 0; e < 8; ++e) pk.h[e] = (__bf16)(bregs[e] * bmask);
    *(v4u*)(smem + 40960 + buf * 5120 + j * 80 + cbase * 2) = pk.q;
  };

  // ---- prologue: stage step 0 into buffer 0 ----
  stageA(0, 0);
  loadB(0);
  storeB(0);
  wait_async0();
  __syncthreads();

  // ---- pipelined K-loop: one barrier per step ----
  for (int step = 0; step < KSTEPS; ++step) {
    const int cur = step & 1;
    const int nxt = cur ^ 1;
    const bool more = (step + 1) < KSTEPS;
    if (more) {
      stageA(step + 1, nxt);   // async, lands in LDS while we compute
      loadB(step + 1);         // loads in flight during WMMAs
    }
    // compute on buffer `cur`
    {
      const __bf16* As = (const __bf16*)(smem + cur * 20480);
      const __bf16* Bs = (const __bf16*)(smem + 40960 + cur * 5120);
      union Frag { v16bf v; v4u q[2]; };
      Frag fb;
      const v4u* pb = (const v4u*)(Bs + (n_sub * 16 + fm) * 40 + khalf * 16);
      fb.q[0] = pb[0]; fb.q[1] = pb[1];
      #pragma unroll
      for (int i = 0; i < 8; ++i) {
        Frag fa;
        int row = (m_half * 8 + i) * 16 + fm;
        const __bf16* pa = As + row * 40 + khalf * 16;
        fa.q[0] = *(const v4u*)pa;
        fa.q[1] = *(const v4u*)(pa + 8);
        acc[i] = __builtin_amdgcn_wmma_f32_16x16x32_bf16(
            false, fa.v, false, fb.v, (short)0, acc[i], false, false);
      }
    }
    if (more) {
      storeB(nxt);             // cvt + single ds_store_b128
      wait_async0();           // our A-tile asyncs for `nxt` done
    }
    __syncthreads();
  }

  // ---- epilogue: bias + ReLU -> LDS transposed [px][ch] (bf16) ----
  __bf16* Ys = (__bf16*)smem;  // [64][264]
  const int pl  = n_sub * 16 + fm;
  const int hi8 = (lane >= 16) ? 8 : 0;
  #pragma unroll
  for (int i = 0; i < 8; ++i) {
    int mbase = (m_half * 8 + i) * 16;
    #pragma unroll
    for (int g = 0; g < 8; ++g) {
      int ch = mbase + g + hi8;
      float y = acc[i][g] + b1[ch];
      Ys[pl * 264 + ch] = (__bf16)fmaxf(y, 0.0f);
    }
  }
  __syncthreads();

  // conv2: 1x1, K=256, WMMA; waves 0..3 each compute one 16x16 tile
  if (wv < 4) {
    v8f a2 = {};
    #pragma unroll
    for (int ks = 0; ks < 8; ++ks) {
      int c0 = ks * 32;
      union Frag { v16bf v; v4u q[2]; };
      Frag fa, fb;
      const __bf16* pw = W2bf + fm * 256 + c0 + khalf * 16;
      fa.q[0] = *(const v4u*)pw;
      fa.q[1] = *(const v4u*)(pw + 8);
      const v4u* pb = (const v4u*)(Ys + (wv * 16 + fm) * 264 + c0 + khalf * 16);
      fb.q[0] = pb[0]; fb.q[1] = pb[1];
      a2 = __builtin_amdgcn_wmma_f32_16x16x32_bf16(
          false, fa.v, false, fb.v, (short)0, a2, false, false);
    }
    #pragma unroll
    for (int g = 0; g < 8; ++g) {
      int k = g + hi8;
      if (k < Kk) {
        int pg = n0 + wv * 16 + fm;
        int bb = pg / HW;
        int hw = pg - bb * HW;
        logits_out[(size_t)bb * (Kk * HW) + k * HW + hw] = a2[g] + b2[k];
      }
    }
  }
}

// ---------------------------------------------------------------------------
// Block reductions (deterministic tree; blockDim == 256)
// ---------------------------------------------------------------------------
__device__ __forceinline__ float blk_sum(float v, float* red) {
  int t = threadIdx.x;
  red[t] = v; __syncthreads();
  for (int s = 128; s > 0; s >>= 1) { if (t < s) red[t] += red[t + s]; __syncthreads(); }
  v = red[0]; __syncthreads(); return v;
}
__device__ __forceinline__ float blk_max(float v, float* red) {
  int t = threadIdx.x;
  red[t] = v; __syncthreads();
  for (int s = 128; s > 0; s >>= 1) { if (t < s) red[t] = fmaxf(red[t], red[t + s]); __syncthreads(); }
  v = red[0]; __syncthreads(); return v;
}

// ---------------------------------------------------------------------------
// Kernel 3: per-batch softmax (tau), cross-K normalize, probs out, coords,
// entropy/diversity partials, bf16 padded probs for the feat_k GEMM.
// ---------------------------------------------------------------------------
__global__ __launch_bounds__(256) void softmax_stats(
    const float* __restrict__ logits, float* __restrict__ out,
    __bf16* __restrict__ probsbf, float* __restrict__ parts)
{
  __shared__ float Ls[Kk * HW];
  __shared__ float red[256];
  __shared__ float norms[Kk];
  const int b = blockIdx.x, t = threadIdx.x;

  for (int i = t; i < Kk * HW; i += 256) Ls[i] = logits[(size_t)b * Kk * HW + i];
  for (int i = t; i < 16 * 800; i += 256) probsbf[(size_t)b * 16 * 800 + i] = (__bf16)0.0f;
  __syncthreads();

  // Per-keypoint tau-softmax over 784 pixels
  for (int k = 0; k < Kk; ++k) {
    float m = -3.4e38f;
    for (int i = t; i < HW; i += 256) m = fmaxf(m, Ls[k * HW + i]);
    m = blk_max(m, red);
    float s = 0.0f;
    for (int i = t; i < HW; i += 256) {
      float e = __expf((Ls[k * HW + i] - m) * (1.0f / TAUf));
      Ls[k * HW + i] = e; s += e;
    }
    s = blk_sum(s, red);
    float inv = 1.0f / s;
    for (int i = t; i < HW; i += 256) Ls[k * HW + i] *= inv;
    __syncthreads();
  }

  // Normalize across keypoints; emit probs (f32 out) and bf16 padded copy
  for (int i = t; i < HW; i += 256) {
    float d = EPSf;
    for (int k = 0; k < Kk; ++k) d += Ls[k * HW + i];
    float invd = 1.0f / d;
    for (int k = 0; k < Kk; ++k) {
      float v = Ls[k * HW + i] * invd;
      Ls[k * HW + i] = v;
      out[OUT_PROBS + (size_t)b * Kk * HW + k * HW + i] = v;
      probsbf[(size_t)b * 16 * 800 + k * 800 + i] = (__bf16)v;
    }
  }
  __syncthreads();

  // Coords (soft-argmax), entropy partial, row norms
  float ent_acc = 0.0f;
  for (int k = 0; k < Kk; ++k) {
    float sx = 0.f, sy = 0.f, he = 0.f, nn = 0.f;
    for (int i = t; i < HW; i += 256) {
      float v  = Ls[k * HW + i];
      float gx = (float)(i % Ww) * (1.0f / 27.0f);
      float gy = (float)(i / Ww) * (1.0f / 27.0f);
      sx += v * gx; sy += v * gy;
      float pc = fmaxf(v, EPSf);
      he -= pc * __logf(pc);
      nn += v * v;
    }
    sx = blk_sum(sx, red); sy = blk_sum(sy, red);
    he = blk_sum(he, red); nn = blk_sum(nn, red);
    if (t == 0) {
      out[OUT_COORD + ((size_t)b * Kk + k) * 2 + 0] = sx;
      out[OUT_COORD + ((size_t)b * Kk + k) * 2 + 1] = sy;
      norms[k] = fmaxf(sqrtf(nn), EPSf);
    }
    ent_acc += he;
  }
  __syncthreads();

  // Diversity partial: sum over k!=j of (cosine sim)^2 (symmetric -> 2x upper)
  float div_acc = 0.0f;
  for (int k = 0; k < Kk; ++k) {
    for (int jj = k + 1; jj < Kk; ++jj) {
      float d = 0.0f;
      for (int i = t; i < HW; i += 256) d += Ls[k * HW + i] * Ls[jj * HW + i];
      d = blk_sum(d, red);
      float c = d / (norms[k] * norms[jj]);
      div_acc += 2.0f * c * c;
    }
  }
  if (t == 0) { parts[b] = div_acc; parts[32 + b] = ent_acc; }
}

// ---------------------------------------------------------------------------
// Kernel 4: feat_k = probs[12x784(pad 16x800)] x feat^T[784x1024] via WMMA
// Grid: 32 batches x 16 channel-tiles (64 ch each); 4 waves, one 16-ch tile ea.
// ---------------------------------------------------------------------------
__global__ __launch_bounds__(128) void featk_gemm(
    const float* __restrict__ feat, const __bf16* __restrict__ probsbf,
    float* __restrict__ out)
{
  const int b    = blockIdx.x >> 4;
  const int ct   = blockIdx.x & 15;
  const int lane = threadIdx.x & 31;
  const int wv   = threadIdx.x >> 5;
  const int fm   = lane & 15;
  const int khalf = lane >> 4;
  const int col  = ct * 64 + wv * 16 + fm;   // channel c
  const float*  fb = feat + ((size_t)b * Cin + col) * HW;
  const __bf16* pa = probsbf + (size_t)b * 16 * 800;

  v8f acc = {};
  for (int ks = 0; ks < 25; ++ks) {
    union Frag { v16bf v; v4u q[2]; };
    Frag a;
    const __bf16* ap = pa + fm * 800 + ks * 32 + khalf * 16;
    a.q[0] = *(const v4u*)ap;
    a.q[1] = *(const v4u*)(ap + 8);

    v16bf bv = {};
    int hw0 = ks * 32 + khalf * 16;
    if (hw0 < HW) {
      v4f f0 = *(const v4f*)(fb + hw0);
      v4f f1 = *(const v4f*)(fb + hw0 + 4);
      v4f f2 = *(const v4f*)(fb + hw0 + 8);
      v4f f3 = *(const v4f*)(fb + hw0 + 12);
      float fv[16] = {f0.x, f0.y, f0.z, f0.w, f1.x, f1.y, f1.z, f1.w,
                      f2.x, f2.y, f2.z, f2.w, f3.x, f3.y, f3.z, f3.w};
      #pragma unroll
      for (int e = 0; e < 16; ++e) bv[e] = (__bf16)fv[e];
    }
    acc = __builtin_amdgcn_wmma_f32_16x16x32_bf16(
        false, a.v, false, bv, (short)0, acc, false, false);
  }
  const int hi8 = (lane >= 16) ? 8 : 0;
  #pragma unroll
  for (int g = 0; g < 8; ++g) {
    int k = g + hi8;
    if (k < Kk)
      out[OUT_FEATK + (size_t)b * Kk * Cin + k * Cin + col] = acc[g] * (1.0f / (float)HW);
  }
}

// ---------------------------------------------------------------------------
// Kernel 5: finalize scalar means (deterministic single-thread sums)
// ---------------------------------------------------------------------------
__global__ void finalize(const float* __restrict__ parts, float* __restrict__ out) {
  if (threadIdx.x == 0 && blockIdx.x == 0) {
    float d = 0.f, e = 0.f;
    for (int b = 0; b < Bn; ++b) { d += parts[b]; e += parts[32 + b]; }
    out[OUT_DIV] = d / (float)(Bn * Kk * Kk);
    out[OUT_ENT] = e / (float)(Bn * Kk);
  }
}

// ---------------------------------------------------------------------------
extern "C" void kernel_launch(void* const* d_in, const int* in_sizes, int n_in,
                              void* d_out, int out_size, void* d_ws, size_t ws_size,
                              hipStream_t stream) {
  const float* feat = (const float*)d_in[0];
  const float* W1   = (const float*)d_in[1];
  const float* b1   = (const float*)d_in[2];
  const float* W2   = (const float*)d_in[3];
  const float* b2   = (const float*)d_in[4];
  float* out = (float*)d_out;
  char*  ws  = (char*)d_ws;

  __bf16* w1bf = (__bf16*)(ws + WS_W1);
  __bf16* w2bf = (__bf16*)(ws + WS_W2);
  float*  lg   = (float*)(ws + WS_LG);
  __bf16* pb   = (__bf16*)(ws + WS_PB);
  float*  pt   = (float*)(ws + WS_PT);

  prep_w1<<<(9 * C1 * Cin + 255) / 256, 256, 0, stream>>>(W1, w1bf);
  prep_w2<<<16, 256, 0, stream>>>(W2, w2bf);
  conv_fused<<<NBLK, 256, 0, stream>>>(feat, w1bf, w2bf, b1, b2, lg);
  softmax_stats<<<Bn, 256, 0, stream>>>(lg, out, pb, pt);
  featk_gemm<<<Bn * 16, 128, 0, stream>>>(feat, pb, out);
  finalize<<<1, 32, 0, stream>>>(pt, out);
}
